// VQTokenizer_84353157693462
// MI455X (gfx1250) — matrix operand order repaired and checked by reading, not compiled, not run on hardware
//
#include <hip/hip_runtime.h>
#include <hip/hip_bf16.h>
#include <math.h>

// ---------------------------------------------------------------------------
// VQ tokenizer forward, fused for MI455X (gfx1250, wave32, WMMA).
//
// Roofline: 270 MB HBM traffic -> ~11.6 us floor @23.3 TB/s; 43 GFLOP total.
// f16 WMMA 16x16x32 (fp32 accum) keeps GEMMs at/below the memory roof; fp32
// WMMA (16x16x4) would be ~8x more matrix instructions and compute-bound.
// ---------------------------------------------------------------------------

typedef _Float16 h4   __attribute__((ext_vector_type(4)));
typedef _Float16 h8   __attribute__((ext_vector_type(8)));
typedef _Float16 v16h __attribute__((ext_vector_type(16)));
typedef float    v8f  __attribute__((ext_vector_type(8)));

#define D_IN   384
#define D_HID  256
#define D_LAT  128
#define NCODES 256
#define ROWS_TOTAL (64 * 2048)   // 131072
#define MTILE  64                // rows per workgroup
#define NWG    (ROWS_TOTAL / MTILE)  // 2048

// LDS strides (halves). Chosen: multiple of 8 halves (16B ds_load_b128
// alignment) and dword-stride % 64 != 0 (bank-conflict-free row stripes).
#define SX_STRIDE 392   // >= 384 ; 392*2/4 = 196 dwords, %64 = 4
#define SH_STRIDE 264   // >= 256 ; 132 dwords, %64 = 4
#define SZ_STRIDE 136   // >= 128 ; 68 dwords,  %64 = 4

__device__ __forceinline__ v8f wmma_f16(v16h a, v16h b, v8f c) {
  // D = A(16x32 f16) x B(32x16 f16) + C(16x16 f32)
  return __builtin_amdgcn_wmma_f32_16x16x32_f16(false, a, false, b,
                                                (short)0, c, false, false);
}

__device__ __forceinline__ v8f vzero8() {
  v8f z;
#pragma unroll
  for (int i = 0; i < 8; ++i) z[i] = 0.0f;
  return z;
}

// Load an A fragment (16x32 f16) for this wave's 16-row stripe from LDS.
// ISA layout: lanes0-15 hold K=kb..kb+7 (halves0-7) and K=kb+16..23 (halves
// 8-15); lanes16-31 hold K=kb+8..15 and K=kb+24..31. Row = lane%16.
__device__ __forceinline__ v16h load_afrag(const _Float16* rowbase, int kb, int lhi) {
  const _Float16* ap = rowbase + kb + lhi * 8;
  h8 alo = *(const h8*)ap;        // ds_load_b128
  h8 ahi = *(const h8*)(ap + 16); // ds_load_b128
  return __builtin_shufflevector(alo, ahi, 0, 1, 2, 3, 4, 5, 6, 7,
                                 8, 9, 10, 11, 12, 13, 14, 15);
}

// ---------------------------------------------------------------------------
// Prep: pack B matrices into WMMA B-fragment order.
// B tile (32x16) fragment: lane n (0-15) holds column n0+n, K = k0+0..7 (halves
// 0-7) and k0+16..23 (halves 8-15); lanes 16-31 hold K = k0+8..15 and k0+24..31.
// One tile = 32 lanes * 32B.
// ---------------------------------------------------------------------------
__global__ void pack_b_kernel(const float* __restrict__ B, _Float16* __restrict__ out,
                              int K, int Nn, int transposed) {
  int tid = blockIdx.x * blockDim.x + threadIdx.x;
  int ntN = Nn >> 4;
  int total = (K >> 5) * ntN * 32;
  if (tid >= total) return;
  int lane = tid & 31;
  int tile = tid >> 5;
  int kt = tile / ntN;
  int nt = tile - kt * ntN;
  int k0 = kt * 32 + ((lane >> 4) << 3);
  int n  = nt * 16 + (lane & 15);
  _Float16* o = out + ((size_t)tile * 32 + lane) * 16;
#pragma unroll
  for (int e = 0; e < 16; ++e) {
    int k = k0 + e + (e >= 8 ? 8 : 0);
    float v = transposed ? B[(size_t)n * K + k] : B[(size_t)k * Nn + n];
    o[e] = (_Float16)v;
  }
}

__global__ void enorm_kernel(const float* __restrict__ cb, float* __restrict__ en) {
  int t = threadIdx.x;  // 256 threads, one per code
  const float* row = cb + (size_t)t * D_LAT;
  float s = 0.0f;
  for (int i = 0; i < D_LAT; ++i) { float v = row[i]; s += v * v; }
  en[t] = s;
}

// ---------------------------------------------------------------------------
// Main fused kernel: 2048 WGs x 256 threads (8 waves). 64 rows per WG.
// Wave w: row stripe (w>>1)*16, column half (w&1).
// ---------------------------------------------------------------------------
__global__ __launch_bounds__(256, 1)
void vq_main_kernel(const float* __restrict__ emb,
                    const _Float16* __restrict__ w1p,
                    const float* __restrict__ b1, const float* __restrict__ g1,
                    const float* __restrict__ be1,
                    const _Float16* __restrict__ w2p,
                    const float* __restrict__ b2, const float* __restrict__ g2,
                    const float* __restrict__ be2,
                    const _Float16* __restrict__ cbp,
                    const float* __restrict__ codebook,
                    const float* __restrict__ enorm2,
                    float* __restrict__ zq_out,
                    int* __restrict__ idx_out,
                    float* __restrict__ partials) {
  __shared__ _Float16 sX[MTILE * SX_STRIDE];  // 50,176 B  (x in f16)
  __shared__ _Float16 sH[MTILE * SH_STRIDE];  // 33,792 B  (h raw -> LN+GELU in place)
  __shared__ _Float16 sZ[MTILE * SZ_STRIDE];  // 17,408 B  (z raw -> LN in place)
  __shared__ float sEn[NCODES];               // codebook row norms
  __shared__ int   sIdx[MTILE];
  __shared__ float sBest[MTILE * 2];
  __shared__ int   sBidx[MTILE * 2];
  __shared__ float sRed[256];

  const int tid  = threadIdx.x;
  const int lane = tid & 31;
  const int wv   = tid >> 5;      // wave 0..7
  const int lrow = lane & 15;     // row within stripe / column within tile
  const int lhi  = lane >> 4;     // lane half
  const int r0   = (wv >> 1) * 16;  // wave's row stripe base (0,16,32,48)
  const int ch   = wv & 1;          // wave's column half
  const size_t rowG0 = (size_t)blockIdx.x * MTILE;

  sEn[tid] = enorm2[tid];

  // ---- Stage X tile: 64 rows x 384 f32 -> f16 in LDS ----
  {
    const float4* src = (const float4*)(emb + rowG0 * D_IN);
    for (int it = 0; it < 24; ++it) {           // 64*96 quads / 256 threads
      int idx = tid + it * 256;
      int row = idx / 96;                       // 96 float4 per row
      int q   = idx - row * 96;
      float4 v = src[(size_t)row * 96 + q];
      h4 hv = { (_Float16)v.x, (_Float16)v.y, (_Float16)v.z, (_Float16)v.w };
      *(h4*)&sX[row * SX_STRIDE + q * 4] = hv;
    }
  }
  __syncthreads();

  // ---- GEMM1: H[64x256] = X @ W1 (raw, bias applied in LN pass) ----
  {
    const _Float16* arow = &sX[(r0 + lrow) * SX_STRIDE];
    for (int ncs = 0; ncs < 2; ++ncs) {
      const int nc = ch * 2 + ncs;              // 64-col chunk, 4 tiles
      v8f acc[4];
#pragma unroll
      for (int t = 0; t < 4; ++t) acc[t] = vzero8();
      for (int k = 0; k < 12; ++k) {            // K = 384
        v16h a = load_afrag(arow, k * 32, lhi);
        const _Float16* bp = w1p + ((size_t)(k * 16 + nc * 4) * 32 + lane) * 16;
#pragma unroll
        for (int t = 0; t < 4; ++t) {
          v16h b = *(const v16h*)(bp + (size_t)t * 512);
          acc[t] = wmma_f16(a, b, acc[t]);
        }
      }
#pragma unroll
      for (int t = 0; t < 4; ++t) {
        int col = nc * 64 + t * 16 + lrow;
#pragma unroll
        for (int j = 0; j < 8; ++j)
          sH[(r0 + j + lhi * 8) * SH_STRIDE + col] = (_Float16)acc[t][j];
      }
    }
  }
  __syncthreads();

  // ---- LN1 + exact GELU (in place). 4 threads per row, 64 cols each ----
  {
    const int row = tid >> 2;
    const int c0  = (tid & 3) * 64;
    _Float16* hr = &sH[row * SH_STRIDE + c0];
    float s = 0.0f, ss = 0.0f;
    for (int i = 0; i < 64; ++i) {
      float x = (float)hr[i] + b1[c0 + i];
      s += x; ss += x * x;
    }
    s  += __shfl_xor(s, 1, 32);  s  += __shfl_xor(s, 2, 32);
    ss += __shfl_xor(ss, 1, 32); ss += __shfl_xor(ss, 2, 32);
    float m    = s * (1.0f / 256.0f);
    float var  = ss * (1.0f / 256.0f) - m * m;
    float rstd = rsqrtf(var + 1e-5f);
    for (int i = 0; i < 64; ++i) {
      int c = c0 + i;
      float x = (float)hr[i] + b1[c];
      float v = (x - m) * rstd * g1[c] + be1[c];
      float ge = 0.5f * v * (1.0f + erff(v * 0.70710678118654752f));
      hr[i] = (_Float16)ge;
    }
  }
  __syncthreads();

  // ---- GEMM2: Z[64x128] = H @ W2 (raw) ----
  {
    const _Float16* arow = &sH[(r0 + lrow) * SH_STRIDE];
    const int nc = ch;                          // one 64-col chunk per wave
    v8f acc[4];
#pragma unroll
    for (int t = 0; t < 4; ++t) acc[t] = vzero8();
    for (int k = 0; k < 8; ++k) {               // K = 256
      v16h a = load_afrag(arow, k * 32, lhi);
      const _Float16* bp = w2p + ((size_t)(k * 8 + nc * 4) * 32 + lane) * 16;
#pragma unroll
      for (int t = 0; t < 4; ++t) {
        v16h b = *(const v16h*)(bp + (size_t)t * 512);
        acc[t] = wmma_f16(a, b, acc[t]);
      }
    }
#pragma unroll
    for (int t = 0; t < 4; ++t) {
      int col = nc * 64 + t * 16 + lrow;
#pragma unroll
      for (int j = 0; j < 8; ++j)
        sZ[(r0 + j + lhi * 8) * SZ_STRIDE + col] = (_Float16)acc[t][j];
    }
  }
  __syncthreads();

  // ---- LN2 (in place -> z in f16). 4 threads per row, 32 cols each ----
  {
    const int row = tid >> 2;
    const int c0  = (tid & 3) * 32;
    _Float16* zr = &sZ[row * SZ_STRIDE + c0];
    float s = 0.0f, ss = 0.0f;
    for (int i = 0; i < 32; ++i) {
      float x = (float)zr[i] + b2[c0 + i];
      s += x; ss += x * x;
    }
    s  += __shfl_xor(s, 1, 32);  s  += __shfl_xor(s, 2, 32);
    ss += __shfl_xor(ss, 1, 32); ss += __shfl_xor(ss, 2, 32);
    float m    = s * (1.0f / 128.0f);
    float var  = ss * (1.0f / 128.0f) - m * m;
    float rstd = rsqrtf(var + 1e-5f);
    for (int i = 0; i < 32; ++i) {
      int c = c0 + i;
      float x = (float)zr[i] + b2[c];
      zr[i] = (_Float16)((x - m) * rstd * g2[c] + be2[c]);
    }
  }
  __syncthreads();

  // ---- GEMM3 + argmin: S = Z @ CB^T; minimize ||e||^2 - 2*S ----
  {
    const _Float16* arow = &sZ[(r0 + lrow) * SZ_STRIDE];
    float best[8];
    int   bidx[8];
#pragma unroll
    for (int j = 0; j < 8; ++j) { best[j] = 3.4e38f; bidx[j] = 0; }
    for (int ncs = 0; ncs < 2; ++ncs) {
      const int nc = ch * 2 + ncs;              // 64 codes per chunk
      v8f acc[4];
#pragma unroll
      for (int t = 0; t < 4; ++t) acc[t] = vzero8();
      for (int k = 0; k < 4; ++k) {             // K = 128
        v16h a = load_afrag(arow, k * 32, lhi);
        const _Float16* bp = cbp + ((size_t)(k * 16 + nc * 4) * 32 + lane) * 16;
#pragma unroll
        for (int t = 0; t < 4; ++t) {
          v16h b = *(const v16h*)(bp + (size_t)t * 512);
          acc[t] = wmma_f16(a, b, acc[t]);
        }
      }
#pragma unroll
      for (int t = 0; t < 4; ++t) {
        int col = nc * 64 + t * 16 + lrow;
        float en = sEn[col];
#pragma unroll
        for (int j = 0; j < 8; ++j) {
          float d = en - 2.0f * acc[t][j];
          if (d < best[j]) { best[j] = d; bidx[j] = col; }
        }
      }
    }
    // reduce across the 16 lanes of each half (columns), ties -> lower index
#pragma unroll
    for (int j = 0; j < 8; ++j) {
#pragma unroll
      for (int off = 1; off < 16; off <<= 1) {
        float ov = __shfl_xor(best[j], off, 32);
        int   oi = __shfl_xor(bidx[j], off, 32);
        if (ov < best[j] || (ov == best[j] && oi < bidx[j])) {
          best[j] = ov; bidx[j] = oi;
        }
      }
    }
    if (lrow == 0) {
#pragma unroll
      for (int j = 0; j < 8; ++j) {
        int row = r0 + lhi * 8 + j;
        sBest[row * 2 + ch] = best[j];
        sBidx[row * 2 + ch] = bidx[j];
      }
    }
  }
  __syncthreads();

  // combine the two column-half candidates per row
  if (tid < MTILE) {
    float v0 = sBest[tid * 2 + 0], v1 = sBest[tid * 2 + 1];
    int   i0 = sBidx[tid * 2 + 0], i1 = sBidx[tid * 2 + 1];
    int win = (v1 < v0 || (v1 == v0 && i1 < i0)) ? i1 : i0;
    sIdx[tid] = win;
    idx_out[rowG0 + tid] = win;  // int32 bit pattern into output slot
  }
  __syncthreads();

  // ---- Output: gather z_q = codebook[idx], accumulate loss partial ----
  {
    const int r   = tid >> 2;
    const int qtr = tid & 3;                    // 32 floats each
    const int code = sIdx[r];
    const float4* cb4 = (const float4*)(codebook + (size_t)code * D_LAT + qtr * 32);
    float4* out4 = (float4*)(zq_out + (rowG0 + r) * D_LAT + qtr * 32);
    const _Float16* zr = &sZ[r * SZ_STRIDE + qtr * 32];
    float lsum = 0.0f;
#pragma unroll
    for (int i = 0; i < 8; ++i) {
      float4 c = cb4[i];
      float d0 = c.x - (float)zr[i * 4 + 0];
      float d1 = c.y - (float)zr[i * 4 + 1];
      float d2 = c.z - (float)zr[i * 4 + 2];
      float d3 = c.w - (float)zr[i * 4 + 3];
      lsum += d0 * d0 + d1 * d1 + d2 * d2 + d3 * d3;
      out4[i] = c;
    }
    sRed[tid] = lsum;
  }
  __syncthreads();
  if (tid == 0) {
    float s = 0.0f;
    for (int i = 0; i < 256; ++i) s += sRed[i];
    partials[blockIdx.x] = s;
  }
}

__global__ void loss_finalize_kernel(const float* __restrict__ partials,
                                     float* __restrict__ loss_out) {
  __shared__ float red[256];
  int t = threadIdx.x;
  float s = 0.0f;
  for (int i = 0; i < 8; ++i) s += partials[t + 256 * i];  // 2048 partials
  red[t] = s;
  __syncthreads();
  for (int off = 128; off > 0; off >>= 1) {
    if (t < off) red[t] += red[t + off];
    __syncthreads();
  }
  // loss = codebook_loss + 0.25*commitment = 1.25 * mean((z_q - z)^2)
  if (t == 0) loss_out[0] = 1.25f * red[0] / (float)(ROWS_TOTAL * D_LAT);
}

// ---------------------------------------------------------------------------
extern "C" void kernel_launch(void* const* d_in, const int* in_sizes, int n_in,
                              void* d_out, int out_size, void* d_ws, size_t ws_size,
                              hipStream_t stream) {
  const float* emb      = (const float*)d_in[0];
  const float* W1       = (const float*)d_in[1];
  const float* b1       = (const float*)d_in[2];
  const float* g1       = (const float*)d_in[3];
  const float* be1      = (const float*)d_in[4];
  const float* W2       = (const float*)d_in[5];
  const float* b2       = (const float*)d_in[6];
  const float* g2       = (const float*)d_in[7];
  const float* be2      = (const float*)d_in[8];
  const float* codebook = (const float*)d_in[9];

  float* out_f = (float*)d_out;
  const size_t ZQ_ELEMS = (size_t)ROWS_TOTAL * D_LAT;  // 16,777,216
  float* d_zq   = out_f;
  float* d_loss = out_f + ZQ_ELEMS;
  int*   d_idx  = (int*)(out_f + ZQ_ELEMS + 1);

  // workspace layout
  _Float16* w1p = (_Float16*)d_ws;        // 384*256 = 98304 halves
  _Float16* w2p = w1p + 98304;            // 256*128 = 32768 halves
  _Float16* cbp = w2p + 32768;            // 128*256 = 32768 halves (CB^T)
  float* enorm2   = (float*)(cbp + 32768);  // 256 floats
  float* partials = enorm2 + 256;           // 2048 floats

  // pack weights into WMMA B-fragment order (tiny; L2-resident thereafter)
  pack_b_kernel<<<24, 256, 0, stream>>>(W1, w1p, D_IN, D_HID, 0);   // 6144 thr
  pack_b_kernel<<<8, 256, 0, stream>>>(W2, w2p, D_HID, D_LAT, 0);   // 2048 thr
  pack_b_kernel<<<8, 256, 0, stream>>>(codebook, cbp, D_LAT, NCODES, 1);
  enorm_kernel<<<1, 256, 0, stream>>>(codebook, enorm2);

  vq_main_kernel<<<NWG, 256, 0, stream>>>(emb, w1p, b1, g1, be1,
                                          w2p, b2, g2, be2,
                                          cbp, codebook, enorm2,
                                          d_zq, d_idx, partials);

  loss_finalize_kernel<<<1, 256, 0, stream>>>(partials, d_loss);
}
